// SSMDiagScan_69011534512413
// MI455X (gfx1250) — compile-verified, matrix operand order checked by reading, not run on hardware
//
#include <hip/hip_runtime.h>
#include <math.h>

// Problem constants (match reference)
#define D_DIM   1024
#define B_DIM   8
#define T_DIM   4096
#define M_DIM   (B_DIM * T_DIM)       // 32768 rows for both GEMMs
#define CHUNK   64
#define NCHUNK  (T_DIM / CHUNK)       // 64 chunks per sequence

typedef __attribute__((ext_vector_type(16))) __bf16 v16bf;
typedef __attribute__((ext_vector_type(8)))  float  v8f;

__device__ __forceinline__ unsigned short f32_to_bf16(float f) {
  unsigned int u = __float_as_uint(f);
  u += 0x7FFFu + ((u >> 16) & 1u);   // round-to-nearest-even
  return (unsigned short)(u >> 16);
}

// ---------------------------------------------------------------------------
// fp32 -> packed bf16 pairs (grid exactly sized; npairs = nelem/2)
// ---------------------------------------------------------------------------
__global__ void cvt_f32_to_bf16_kernel(const float* __restrict__ in,
                                       unsigned int* __restrict__ out,
                                       int npairs) {
  int i = blockIdx.x * blockDim.x + threadIdx.x;
  if (i < npairs) {
    float2 f = ((const float2*)in)[i];
    out[i] = (unsigned int)f32_to_bf16(f.x) |
             ((unsigned int)f32_to_bf16(f.y) << 16);
  }
}

// ---------------------------------------------------------------------------
// C[m,n] = sum_k A[m,k] * Bw[n,k]   (A: 32768 x 1024 bf16, Bw: 1024 x 1024 bf16)
//
// Block = 8 waves = 256x64 output tile, all waves share one 64-column group.
// Per K-step (32): the block stages the shared B slab (64 cols x 32 k bf16 =
// 4 KB) into LDS with one global_load_async_to_lds_b128 per thread (ASYNCcnt,
// double-buffered), waves pull their private A rows via global_load_b128, and
// each wave runs 8 x v_wmma_f32_16x16x32_bf16 (2 m-tiles x 4 n-tiles) off
// ds_load_b128 reads of the staged slab.
//
// 16-bit A-frag layout (ISA 7.12.2): lane={half,m%16}; per lane two 16B runs
// at k0+half*8 and k0+16+half*8. B (=W^T) mirrored with lane%16 = column n.
// C/D layout: VGPR r -> row 16*tile + half*8 + r, col = lane%16.
// ---------------------------------------------------------------------------
__global__ __launch_bounds__(256)
void gemm_bf16_wmma_kernel(const unsigned short* __restrict__ A,
                           const unsigned short* __restrict__ Bw,
                           float* __restrict__ C) {
  const int K = D_DIM, N = D_DIM;
  __shared__ unsigned short smem[2 * 64 * 32];   // double buffer, 4 KB each

  const int tid  = threadIdx.x;
  const int lane = tid & 31;
  const int wave = tid >> 5;
  const int half = lane >> 4;
  const int l16  = lane & 15;

  const int nGroup  = blockIdx.x & 15;     // 16 column groups of 64
  const int mBlock  = blockIdx.x >> 4;     // 128 row blocks of 256
  const int colBase = nGroup * 64;
  const int rowBase = mBlock * 256 + wave * 32;

  // Staging assignment: thread -> (column, 16B chunk) of the 4 KB slab
  const int sCol   = tid >> 2;             // 0..63
  const int sChunk = tid & 3;              // 0..3
  const unsigned long long bBase = (unsigned long long)Bw;
  const unsigned sVoffBase = (unsigned)((colBase + sCol) * (K * 2) + sChunk * 16);
  const unsigned sLdsBase  = (unsigned)(uintptr_t)&smem[sCol * 32 + sChunk * 8];

  // Prestage k0 = 0 into buffer 0 (GVS form: lds_vdst, voffset, saddr)
  {
    unsigned voff = sVoffBase;
    unsigned ldso = sLdsBase;
    asm volatile("global_load_async_to_lds_b128 %0, %1, %2"
                 :: "v"(ldso), "v"(voff), "s"(bBase) : "memory");
  }

  const unsigned int* __restrict__ Arow0 =
      (const unsigned int*)(A + (size_t)(rowBase + l16) * K);
  const unsigned int* __restrict__ Arow1 =
      (const unsigned int*)(A + (size_t)(rowBase + 16 + l16) * K);

  union Frag { v16bf v; uint4 q[2]; };
  v8f acc[2][4] = {};

  for (int k0 = 0; k0 < K; k0 += 32) {
    // Producer->consumer edge for the slab staged last iteration.
    asm volatile("s_wait_asynccnt 0x0" ::: "memory");
    __syncthreads();

    const int cur = (k0 >> 5) & 1;

    // Keep one slab in flight: stage next K-step into the other buffer.
    if (k0 + 32 < K) {
      unsigned voff = sVoffBase + (unsigned)((k0 + 32) * 2);
      unsigned ldso = sLdsBase + (unsigned)((cur ^ 1) * 64 * 32 * 2);
      asm volatile("global_load_async_to_lds_b128 %0, %1, %2"
                   :: "v"(ldso), "v"(voff), "s"(bBase) : "memory");
    }

    // Private A fragments (two 16-row tiles), 2x b128 each.
    Frag a0, a1;
    {
      const uint4* p0 = (const uint4*)(Arow0 + ((k0 + half * 8) >> 1));
      a0.q[0] = p0[0];
      a0.q[1] = p0[2];                      // +16 halves = +2 uint4
      const uint4* p1 = (const uint4*)(Arow1 + ((k0 + half * 8) >> 1));
      a1.q[0] = p1[0];
      a1.q[1] = p1[2];
    }

    // Consume staged B slab from LDS; each B frag feeds two WMMAs.
    const unsigned short* sb = &smem[cur * (64 * 32)];
#pragma unroll
    for (int t = 0; t < 4; ++t) {
      Frag b;
      const uint4* q = (const uint4*)&sb[(t * 16 + l16) * 32 + half * 8];
      b.q[0] = q[0];
      b.q[1] = q[2];                        // +16 halves = +2 uint4
      acc[0][t] = __builtin_amdgcn_wmma_f32_16x16x32_bf16(
          false, a0.v, false, b.v, (short)0, acc[0][t], false, false);
      acc[1][t] = __builtin_amdgcn_wmma_f32_16x16x32_bf16(
          false, a1.v, false, b.v, (short)0, acc[1][t], false, false);
    }
  }

#pragma unroll
  for (int mt = 0; mt < 2; ++mt) {
#pragma unroll
    for (int t = 0; t < 4; ++t) {
      int col = colBase + t * 16 + l16;
#pragma unroll
      for (int r = 0; r < 8; ++r) {
        int row = rowBase + mt * 16 + half * 8 + r;
        C[(size_t)row * N + col] = acc[mt][t][r];
      }
    }
  }
}

// ---------------------------------------------------------------------------
// Scan phase A: per-(b, chunk, d) local scan with zero init (in-place on u),
// chunk-end value written to carry[b, c, d]. 512K independent threads.
// ---------------------------------------------------------------------------
__global__ void scan_local_kernel(float* __restrict__ u,
                                  float* __restrict__ carry,
                                  const float* __restrict__ raw_a) {
  int tid = blockIdx.x * blockDim.x + threadIdx.x;   // B*NCHUNK*D threads
  int d = tid % D_DIM;
  int c = (tid / D_DIM) % NCHUNK;
  int b = tid / (D_DIM * NCHUNK);
  float a = tanhf(raw_a[d]);
  size_t base = ((size_t)(b * T_DIM + c * CHUNK)) * D_DIM + d;
  float h = 0.f;
  for (int j = 0; j < CHUNK; ++j) {
    size_t idx = base + (size_t)j * D_DIM;
    h = a * h + u[idx];
    u[idx] = h;                      // store local-scan value
  }
  carry[((size_t)(b * NCHUNK + c)) * D_DIM + d] = h;
}

// ---------------------------------------------------------------------------
// Scan phase B: per-(b, d) sequential combine of NCHUNK chunk carries.
// cin[b, c, d] = state entering chunk c; a^CHUNK via 6 squarings.
// ---------------------------------------------------------------------------
__global__ void scan_carry_kernel(const float* __restrict__ carry,
                                  float* __restrict__ cin,
                                  const float* __restrict__ raw_a,
                                  const float* __restrict__ h0) {
  int tid = blockIdx.x * blockDim.x + threadIdx.x;   // B*D threads
  int d = tid % D_DIM;
  int b = tid / D_DIM;
  float a = tanhf(raw_a[d]);
  float aL = a;
#pragma unroll
  for (int i = 0; i < 6; ++i) aL *= aL;              // a^64
  float h = h0[(size_t)b * D_DIM + d];               // state before t=0
  for (int c = 0; c < NCHUNK; ++c) {
    size_t idx = ((size_t)(b * NCHUNK + c)) * D_DIM + d;
    cin[idx] = h;
    h = aL * h + carry[idx];
  }
}

// ---------------------------------------------------------------------------
// Scan phase C: traj = local + a^{j+1} * cin; emit bf16 traj for GEMM2 and
// h_last (fp32) for the final timestep.
// ---------------------------------------------------------------------------
__global__ void scan_fixup_kernel(const float* __restrict__ ulocal,
                                  const float* __restrict__ cin,
                                  const float* __restrict__ raw_a,
                                  unsigned short* __restrict__ trajb,
                                  float* __restrict__ hlast) {
  int tid = blockIdx.x * blockDim.x + threadIdx.x;   // B*NCHUNK*D threads
  int d = tid % D_DIM;
  int c = (tid / D_DIM) % NCHUNK;
  int b = tid / (D_DIM * NCHUNK);
  float a = tanhf(raw_a[d]);
  float ci = cin[((size_t)(b * NCHUNK + c)) * D_DIM + d];
  size_t base = ((size_t)(b * T_DIM + c * CHUNK)) * D_DIM + d;
  float p = a;
  float val = 0.f;
  for (int j = 0; j < CHUNK; ++j) {
    size_t idx = base + (size_t)j * D_DIM;
    val = ulocal[idx] + p * ci;
    trajb[idx] = f32_to_bf16(val);
    p *= a;
  }
  if (c == NCHUNK - 1) hlast[(size_t)b * D_DIM + d] = val;
}

// ---------------------------------------------------------------------------
// Launch: cvt(x, W_in, W_out) -> GEMM1 -> scanA -> scanB -> scanC -> GEMM2
// ---------------------------------------------------------------------------
extern "C" void kernel_launch(void* const* d_in, const int* in_sizes, int n_in,
                              void* d_out, int out_size, void* d_ws, size_t ws_size,
                              hipStream_t stream) {
  const float* x     = (const float*)d_in[0];   // (B,T,D)
  const float* h0    = (const float*)d_in[1];   // (B,D)
  const float* raw_a = (const float*)d_in[2];   // (D,)
  const float* W_in  = (const float*)d_in[3];   // (D,D)
  const float* W_out = (const float*)d_in[4];   // (D,D)

  float* out   = (float*)d_out;                       // (B,T,D) flat
  float* hlast = out + (size_t)M_DIM * D_DIM;         // then (B,D)

  // Workspace carve-up (~264 MiB total)
  unsigned short* xb    = (unsigned short*)d_ws;                       // M*D bf16
  unsigned short* winb  = xb + (size_t)M_DIM * D_DIM;                  // D*D bf16
  unsigned short* woutb = winb + (size_t)D_DIM * D_DIM;                // D*D bf16
  float* u     = (float*)(woutb + (size_t)D_DIM * D_DIM);              // M*D f32
  float* carry = u + (size_t)M_DIM * D_DIM;                            // B*NC*D f32
  float* cin   = carry + (size_t)B_DIM * NCHUNK * D_DIM;               // B*NC*D f32
  unsigned short* trajb = (unsigned short*)(cin + (size_t)B_DIM * NCHUNK * D_DIM);

  const int threads = 256;

  // bf16 conversions
  {
    int npairs = (M_DIM * D_DIM) / 2;                 // 16,777,216
    cvt_f32_to_bf16_kernel<<<npairs / threads, threads, 0, stream>>>(
        x, (unsigned int*)xb, npairs);
  }
  {
    int npairs = (D_DIM * D_DIM) / 2;                 // 524,288
    cvt_f32_to_bf16_kernel<<<npairs / threads, threads, 0, stream>>>(
        W_in, (unsigned int*)winb, npairs);
    cvt_f32_to_bf16_kernel<<<npairs / threads, threads, 0, stream>>>(
        W_out, (unsigned int*)woutb, npairs);
  }

  // GEMM1: u = x @ W_in^T   (block = 256x64 tile -> 128 * 16 = 2048 blocks)
  {
    int blocks = (M_DIM / 256) * (D_DIM / 64);        // 2048
    gemm_bf16_wmma_kernel<<<blocks, threads, 0, stream>>>(xb, winb, u);
  }

  // Chunked diagonal scan
  {
    int n = B_DIM * NCHUNK * D_DIM;                   // 524,288
    scan_local_kernel<<<n / threads, threads, 0, stream>>>(u, carry, raw_a);
    scan_carry_kernel<<<(B_DIM * D_DIM) / threads, threads, 0, stream>>>(
        carry, cin, raw_a, h0);
    scan_fixup_kernel<<<n / threads, threads, 0, stream>>>(
        u, cin, raw_a, trajb, hlast);
  }

  // GEMM2: out = traj @ W_out^T
  {
    int blocks = (M_DIM / 256) * (D_DIM / 64);
    gemm_bf16_wmma_kernel<<<blocks, threads, 0, stream>>>(trajb, woutb, out);
  }
}
